// CalculateVector_62801011802525
// MI455X (gfx1250) — compile-verified, matrix operand order checked by reading, not run on hardware
//
#include <hip/hip_runtime.h>

// CDNA5 / gfx1250, wave32. One wave per pixel, v_wmma_f32_16x16x32_f16 for the
// K=16 SAD reduction with the spiral tie-break folded into the WMMA accumulator.

typedef __attribute__((ext_vector_type(16))) _Float16 v16h;
typedef __attribute__((ext_vector_type(8)))  float    v8f;
typedef __attribute__((ext_vector_type(4)))  float    v4f;

#define NPIX 32400   // 1*135*240
#define NCH  82      // (2*4+1)^2 + 1
#define NBM  81      // channels participating in spiral argmin

struct alignas(16) Tables {
  float fracC[96];   // spiral[c]/128 for c<81, else 1e30 (pad / input-MV excluded)
  int   inv[82];     // inv[spiral_value] = channel  (spiral is a bijection 1..81)
  int   pad[2];
};

static constexpr Tables make_tables() {
  Tables t{};
  int sp[81] = {};
  int n = 0;
  const int R = 4;
  for (int dy = 0; dy <= R; ++dy) {
    for (int dir = 0; dir < 4; ++dir) {
      const int limit = (dir == 0 && dy == 0) ? 1 : dy;
      for (int dx = -dy; dx < limit; ++dx) {
        int wx = 0, wy = 0;
        if (dir == 0)      { wx = R + dx; wy = R + dy; }
        else if (dir == 1) { wx = R + dy; wy = R - dx; }
        else if (dir == 2) { wx = R - dx; wy = R - dy; }
        else               { wx = R - dy; wy = R + dx; }
        ++n;
        sp[wy * 9 + wx] = n;
      }
    }
  }
  for (int c = 0; c < 96; ++c)
    t.fracC[c] = (c < NBM) ? (float)sp[c] * (1.0f / 128.0f) : 1.0e30f;
  for (int c = 0; c < NBM; ++c) t.inv[sp[c]] = c;
  return t;
}

__constant__ Tables g_tab = make_tables();

__global__ __launch_bounds__(256)
void bm_sad_wmma_kernel(const float* __restrict__ win,
                        const float* __restrict__ tmpl,
                        float* __restrict__ out_vec,
                        float* __restrict__ out_tmpl,
                        float* __restrict__ out_mask,
                        float* __restrict__ out_cost)
{
  const int lane = (int)(threadIdx.x & 31u);
  const int p    = (int)blockIdx.x * ((int)blockDim.x >> 5) + ((int)threadIdx.x >> 5);
  if (p >= NPIX) return;                 // uniform per wave; EXEC stays all-ones

  const int m  = lane & 15;              // WMMA row (channel within group of 16)
  const int hi = lane >> 4;              // which 8-wide K half this lane supplies

  // Template: 8 floats per lane (halves split across lane groups).
  const v4f* tp = (const v4f*)(tmpl + ((size_t)p << 4) + (hi << 3));
  const v4f t0 = tp[0];
  const v4f t1 = tp[1];

  v16h bones;                            // all-ones B matrix (layout-invariant)
#pragma unroll
  for (int i = 0; i < 16; ++i) bones[i] = (_Float16)1.0f;

  float best = 1.0e30f;
  float sum8 = 0.0f;                     // channel-81 partial SAD (lanes 1 and 17)

#pragma unroll
  for (int w = 0; w < 6; ++w) {
    const int cA = 16 * w + m;           // this lane's channel for this WMMA
    float dif[8] = {0.f, 0.f, 0.f, 0.f, 0.f, 0.f, 0.f, 0.f};
    if (w < 5 || m < 2) {                // channels 82..95 are padding: no load
      const v4f* wp =
          (const v4f*)(win + (((size_t)p * NCH + (size_t)cA) << 4) + (hi << 3));
      const v4f a0 = __builtin_nontemporal_load(wp);      // 170MB stream, read once
      const v4f a1 = __builtin_nontemporal_load(wp + 1);
      dif[0] = fabsf(a0.x - t0.x);
      dif[1] = fabsf(a0.y - t0.y);
      dif[2] = fabsf(a0.z - t0.z);
      dif[3] = fabsf(a0.w - t0.w);
      dif[4] = fabsf(a1.x - t1.x);
      dif[5] = fabsf(a1.y - t1.y);
      dif[6] = fabsf(a1.z - t1.z);
      dif[7] = fabsf(a1.w - t1.w);
    }
    if (w == 5) {                        // capture channel-81 cost on lanes 1/17
      sum8 = ((dif[0] + dif[1]) + (dif[2] + dif[3])) +
             ((dif[4] + dif[5]) + (dif[6] + dif[7]));
    }

    // A fragment: 8 exact-integer f16 diffs in K-halves 0..7, zeros in K16..31.
    v16h a;
#pragma unroll
    for (int j = 0; j < 8; ++j) a[j] = (_Float16)dif[j];
#pragma unroll
    for (int j = 8; j < 16; ++j) a[j] = (_Float16)0.0f;

    // C init = spiral tie-break fraction per channel (matches D layout M=v+8*hi).
    const float* cfp = g_tab.fracC + (16 * w + (hi << 3));
    v8f cin;
#pragma unroll
    for (int v = 0; v < 8; ++v) cin[v] = cfp[v];

    // D[m,n] = SAD(chan 16w+m) + spiral(chan)/128, exact in f32.
    const v8f dd = __builtin_amdgcn_wmma_f32_16x16x32_f16(
        false, a, false, bones, (short)0, cin, false, false);

#pragma unroll
    for (int v = 0; v < 8; ++v) best = fminf(best, dd[v]);
  }

  // Wave-wide min of (cost + spiral/128) keys.
#pragma unroll
  for (int off = 16; off >= 1; off >>= 1)
    best = fminf(best, __shfl_xor(best, off, 32));

  // Channel-81 (input MV) cost: halves live on lanes 1 (k0..7) and 17 (k8..15).
  const float c81f = __shfl(sum8, 1, 32) + __shfl(sum8, 17, 32);

  const float costf   = floorf(best);
  const int   sidx    = (int)((best - costf) * 128.0f + 0.5f);  // exact spiral id
  const int   idx_bm  = g_tab.inv[sidx];
  const int   cost_bm = (int)costf;
  const int   c81     = (int)c81f;

  const bool  mv      = (c81 < cost_bm);         // strict <, as in reference
  const int   min_idx = mv ? NBM : idx_bm;
  const int   min_c   = mv ? c81 : cost_bm;

  // min_templates gather: 16 lanes copy the winning window.
  if (lane < 16) {
    out_tmpl[((size_t)p << 4) + lane] =
        win[(((size_t)p * NCH + (size_t)min_idx) << 4) + lane];
  }
  if (lane == 0) {
    out_vec[2 * (size_t)p]     = (float)(4 - idx_bm / 9);   // -(jj) = 4 - row
    out_vec[2 * (size_t)p + 1] = (float)(4 - idx_bm % 9);   // -(ii) = 4 - col
    out_mask[p] = mv ? 1.0f : 0.0f;
    out_cost[p] = (float)min_c;
  }
}

extern "C" void kernel_launch(void* const* d_in, const int* in_sizes, int n_in,
                              void* d_out, int out_size, void* d_ws, size_t ws_size,
                              hipStream_t stream) {
  (void)in_sizes; (void)n_in; (void)d_ws; (void)ws_size; (void)out_size;
  const float* win  = (const float*)d_in[0];
  const float* tmpl = (const float*)d_in[1];
  // d_in[2] = search_range (compile-time 4 here).
  float* out      = (float*)d_out;
  float* out_vec  = out;                              // NPIX*2  (vector, fp16 values)
  float* out_tmpl = out + (size_t)NPIX * 2;           // NPIX*16 (min_templates)
  float* out_mask = out_tmpl + (size_t)NPIX * 16;     // NPIX    (input_mv_mask)
  float* out_cost = out_mask + NPIX;                  // NPIX    (min_cost, int values)

  const int waves_per_block = 256 / 32;               // wave32: 8 waves per block
  const int blocks = (NPIX + waves_per_block - 1) / waves_per_block;  // 4050
  bm_sad_wmma_kernel<<<blocks, 256, 0, stream>>>(win, tmpl, out_vec, out_tmpl,
                                                 out_mask, out_cost);
}
